// CustomEncoderWithAction_76802605187721
// MI455X (gfx1250) — matrix operand order repaired and checked by reading, not compiled, not run on hardware
//
#include <hip/hip_runtime.h>

typedef __attribute__((ext_vector_type(16))) _Float16 v16h;
typedef __attribute__((ext_vector_type(8)))  float    v8f;

#define NAGENTS 1536
#define TSTEPS  8
#define NROB    192

__device__ __forceinline__ unsigned pack_f16x2(float a, float b) {
  union { _Float16 h[2]; unsigned u; } p;
  p.h[0] = (_Float16)a; p.h[1] = (_Float16)b;
  return p.u;
}
__device__ __forceinline__ float sigmoidf_(float x) { return 1.0f / (1.0f + __expf(-x)); }

// ---------------------------------------------------------------------------
// Kernel 1: Hist_Encoder (embed + 8-step LSTM), one thread per agent.
// Writes enc_hist as f16 [N,16] into workspace.
// ---------------------------------------------------------------------------
__global__ __launch_bounds__(256) void lstm_encoder_kernel(
    const float* __restrict__ traj_rel,
    const float* __restrict__ W_he, const float* __restrict__ b_he,
    const float* __restrict__ W_ih, const float* __restrict__ W_hh,
    const float* __restrict__ b_ih, const float* __restrict__ b_hh,
    _Float16* __restrict__ hist_f16)
{
  __shared__ float sWih[16 * 64];
  __shared__ float sWhh[16 * 64];
  __shared__ float sWhe[32];
  __shared__ float sBg[64];
  __shared__ float sBe[16];
  int tid = threadIdx.x;
  for (int k = tid; k < 16 * 64; k += 256) { sWih[k] = W_ih[k]; sWhh[k] = W_hh[k]; }
  if (tid < 32) sWhe[tid] = W_he[tid];
  if (tid < 64) sBg[tid] = b_ih[tid] + b_hh[tid];
  if (tid < 16) sBe[tid] = b_he[tid];
  __syncthreads();

  int n = blockIdx.x * 256 + tid;   // grid is exactly N/256 blocks
  float h[16], c[16];
  #pragma unroll
  for (int k = 0; k < 16; ++k) { h[k] = 0.f; c[k] = 0.f; }

  for (int t = 0; t < TSTEPS; ++t) {
    float x0 = traj_rel[(t * NAGENTS + n) * 2 + 0];
    float x1 = traj_rel[(t * NAGENTS + n) * 2 + 1];
    float e[16];
    #pragma unroll
    for (int k = 0; k < 16; ++k) e[k] = sBe[k] + x0 * sWhe[k] + x1 * sWhe[16 + k];

    float gi[16], gf[16], gg[16], go[16];
    for (int q = 0; q < 16; ++q) {
      float si = sBg[q], sf2 = sBg[16 + q], sg = sBg[32 + q], so = sBg[48 + q];
      #pragma unroll
      for (int k = 0; k < 16; ++k) {
        float a = e[k], b = h[k];
        si  += a * sWih[k * 64 + q]        + b * sWhh[k * 64 + q];
        sf2 += a * sWih[k * 64 + 16 + q]   + b * sWhh[k * 64 + 16 + q];
        sg  += a * sWih[k * 64 + 32 + q]   + b * sWhh[k * 64 + 32 + q];
        so  += a * sWih[k * 64 + 48 + q]   + b * sWhh[k * 64 + 48 + q];
      }
      gi[q] = si; gf[q] = sf2; gg[q] = sg; go[q] = so;
    }
    #pragma unroll
    for (int k = 0; k < 16; ++k) {
      float ik = sigmoidf_(gi[k]);
      float fk = sigmoidf_(gf[k]);
      float gk = tanhf(gg[k]);
      float ok = sigmoidf_(go[k]);
      c[k] = fk * c[k] + ik * gk;
      h[k] = ok * tanhf(c[k]);
    }
  }
  #pragma unroll
  for (int k = 0; k < 16; ++k) hist_f16[n * 16 + k] = (_Float16)h[k];
}

// ---------------------------------------------------------------------------
// Kernel 2: Pooling net via chained WMMA. One wave per query agent i.
// GEMM1: u1^T = W_p1^T(64x32) x u^T(32x16j)  -> 4x wmma f32_16x16x32_f16
// GEMM2: u2^T = W_p2^T(16x64) x relu(u1^T)   -> 2x wmma (K split 32+32)
// j rides the N/lane axis, so masked max-pool is per-lane cndmask+max.
// ---------------------------------------------------------------------------
__global__ __launch_bounds__(256) void pooling_wmma_kernel(
    const float* __restrict__ pos_last,      // [N,2] last-frame positions (16B aligned)
    const _Float16* __restrict__ hist_f16,   // [N,16]
    const int* __restrict__ neigh,           // [N,N]
    const float* __restrict__ W_sp, const float* __restrict__ b_sp,
    const float* __restrict__ W_p1, const float* __restrict__ b_p1,
    const float* __restrict__ W_p2, const float* __restrict__ b_p2,
    float* __restrict__ pooled)              // [N,16]
{
  __shared__ float    s_pos[NAGENTS * 2];     // 12 KB
  __shared__ _Float16 s_hist[NAGENTS * 16];   // 48 KB (320KB LDS per WGP)
  int tid = threadIdx.x;
  // Vectorized staging: B128 global loads -> B128 LDS stores.
  {
    const uint4* gp = (const uint4*)pos_last;
    uint4*       sp = (uint4*)s_pos;
    #pragma unroll
    for (int k = 0; k < (NAGENTS * 2 * 4) / 16 / 256; ++k)   // 3 iters
      sp[tid + k * 256] = gp[tid + k * 256];
    const uint4* gh = (const uint4*)hist_f16;
    uint4*       sh = (uint4*)s_hist;
    #pragma unroll
    for (int k = 0; k < (NAGENTS * 16 * 2) / 16 / 256; ++k)  // 12 iters
      sh[tid + k * 256] = gh[tid + k * 256];
  }
  __syncthreads();

  const int  lane = tid & 31;
  const int  wave = tid >> 5;
  const int  jl   = lane & 15;
  const bool hi   = lane >= 16;
  const int  i    = blockIdx.x * 8 + wave;    // 192 blocks * 8 waves = 1536

  // A-fragment K-index map (contiguous, matching B):
  //   lane<16 : k=s      (K 0..15)
  //   lane>=16: k=s+16   (K 16..31)
  v16h A1[4]; v16h A2a, A2b;
  #pragma unroll
  for (int s = 0; s < 16; ++s) {
    int k = s + (hi ? 16 : 0);
    #pragma unroll
    for (int mb = 0; mb < 4; ++mb)
      A1[mb][s] = (_Float16)W_p1[k * 64 + mb * 16 + jl];   // W_p1^T[row=out, k]
    A2a[s] = (_Float16)W_p2[k * 16 + jl];                   // W_p2^T, K 0..31
    A2b[s] = (_Float16)W_p2[(k + 32) * 16 + jl];            // W_p2^T, K 32..63
  }
  // Bias pre-baked into C accumulators (C/D layout: elem g -> row g + 8*hi)
  v8f C1b[4]; v8f C2b;
  #pragma unroll
  for (int g = 0; g < 8; ++g) {
    int row = g + (hi ? 8 : 0);
    #pragma unroll
    for (int mb = 0; mb < 4; ++mb) C1b[mb][g] = b_p1[mb * 16 + row];
    C2b[g] = b_p2[row];
  }
  // Spatial-embedding weights: uniform addresses -> SGPR-resident
  float w0[16], w1[16], bsv[16];
  #pragma unroll
  for (int e = 0; e < 16; ++e) { w0[e] = W_sp[e]; w1[e] = W_sp[16 + e]; bsv[e] = b_sp[e]; }

  const float pix = s_pos[2 * i], piy = s_pos[2 * i + 1];
  const int* nrow = neigh + i * NAGENTS;
  float acc[8];
  #pragma unroll
  for (int g = 0; g < 8; ++g) acc[g] = 0.f;

  for (int j0 = 0; j0 < NAGENTS; j0 += 16) {
    const int j  = j0 + jl;
    const int ng = nrow[j];
    // Prefetch the mask row a few chunks ahead (wraps; speculative is safe).
    __builtin_prefetch(&nrow[(j0 + 64) & (NAGENTS - 1)], 0, 1);
    const float dx = pix - s_pos[2 * j];
    const float dy = piy - s_pos[2 * j + 1];

    // B1 = u^T: lanes 0..15 carry K0..15 = spatial embed r_ij(j=jl),
    //           lanes 16..31 carry K16..31 = h_j (32B LDS read).
    union { v16h v; unsigned u[8]; } B1;
    const unsigned* hp = (const unsigned*)(s_hist + j * 16);
    #pragma unroll
    for (int q = 0; q < 8; ++q) {
      float r0 = bsv[2 * q]     + dx * w0[2 * q]     + dy * w1[2 * q];
      float r1 = bsv[2 * q + 1] + dx * w0[2 * q + 1] + dy * w1[2 * q + 1];
      unsigned rw = pack_f16x2(r0, r1);
      B1.u[q] = hi ? hp[q] : rw;
    }

    v8f d1[4];
    #pragma unroll
    for (int mb = 0; mb < 4; ++mb)
      d1[mb] = __builtin_amdgcn_wmma_f32_16x16x32_f16(
          false, A1[mb], false, B1.v, (short)0, C1b[mb], false, false);

    // relu + pack to f16 pairs (block mb holds feats mb*16 + g + 8*hi)
    unsigned tb[4][4];
    #pragma unroll
    for (int mb = 0; mb < 4; ++mb)
      #pragma unroll
      for (int q = 0; q < 4; ++q)
        tb[mb][q] = pack_f16x2(fmaxf(d1[mb][2 * q], 0.f), fmaxf(d1[mb][2 * q + 1], 0.f));

    // D-layout -> B-layout fix-up: half-wave swap via shfl_xor(16)
    union { v16h v; unsigned u[8]; } B2a, B2b;
    #pragma unroll
    for (int q = 0; q < 4; ++q) {
      unsigned x0 = __shfl_xor(tb[0][q], 16, 32);
      unsigned x1 = __shfl_xor(tb[1][q], 16, 32);
      unsigned x2 = __shfl_xor(tb[2][q], 16, 32);
      unsigned x3 = __shfl_xor(tb[3][q], 16, 32);
      B2a.u[q]     = hi ? x1        : tb[0][q];
      B2a.u[4 + q] = hi ? tb[1][q]  : x0;
      B2b.u[q]     = hi ? x3        : tb[2][q];
      B2b.u[4 + q] = hi ? tb[3][q]  : x2;
    }

    v8f d2 = __builtin_amdgcn_wmma_f32_16x16x32_f16(
        false, A2a, false, B2a.v, (short)0, C2b, false, false);
    d2 = __builtin_amdgcn_wmma_f32_16x16x32_f16(
        false, A2b, false, B2b.v, (short)0, d2, false, false);

    // masked max-pool: neighbor mask is per-column(j) = per-lane
    #pragma unroll
    for (int g = 0; g < 8; ++g) {
      float u = fmaxf(d2[g], 0.f);           // second relu
      acc[g] = (ng > 0) ? fmaxf(acc[g], u) : acc[g];
    }
  }

  // max across the 16 column slots inside each half-wave
  #pragma unroll
  for (int d = 1; d < 16; d <<= 1) {
    #pragma unroll
    for (int g = 0; g < 8; ++g)
      acc[g] = fmaxf(acc[g], __shfl_xor(acc[g], d, 32));
  }
  if (jl == 0) {
    float* o = pooled + i * 16 + (hi ? 8 : 0);
    #pragma unroll
    for (int g = 0; g < 8; ++g) o[g] = acc[g];
  }
}

// ---------------------------------------------------------------------------
// Kernel 3: fusion MLP. One block per robot, one thread per output feature.
// ---------------------------------------------------------------------------
__global__ __launch_bounds__(256) void fusion_kernel(
    const float* __restrict__ pos_last,
    const int* __restrict__ robot_idx,
    const float* __restrict__ r_goal,
    const float* __restrict__ r_pose,
    const float* __restrict__ action,
    const float* __restrict__ pooled,
    const float* __restrict__ W_emb, const float* __restrict__ b_emb,
    const float* __restrict__ W_fc,  const float* __restrict__ b_fc,
    float* __restrict__ out)
{
  __shared__ float sf[40];
  int b = blockIdx.x, t = threadIdx.x;
  if (t < 16) {
    int r = robot_idx[b];
    float i0 = r_goal[b * 2]     - pos_last[r * 2];
    float i1 = r_goal[b * 2 + 1] - pos_last[r * 2 + 1];
    float a0 = action[b * 2], a1 = action[b * 2 + 1];
    float v = b_emb[t] + i0 * W_emb[t] + i1 * W_emb[16 + t]
                       + a0 * W_emb[32 + t] + a1 * W_emb[48 + t];
    sf[t]      = fmaxf(v, 0.f);
    sf[16 + t] = pooled[r * 16 + t];
  }
  if (t < 5) sf[32 + t] = r_pose[b * 5 + t];
  __syncthreads();
  float s = b_fc[t];
  #pragma unroll
  for (int k = 0; k < 37; ++k) s += sf[k] * W_fc[k * 256 + t];
  out[b * 256 + t] = fmaxf(s, 0.f);
}

// ---------------------------------------------------------------------------
extern "C" void kernel_launch(void* const* d_in, const int* in_sizes, int n_in,
                              void* d_out, int out_size, void* d_ws, size_t ws_size,
                              hipStream_t stream) {
  (void)in_sizes; (void)n_in; (void)out_size; (void)ws_size;
  const float* obs_traj_pos = (const float*)d_in[0];
  const float* traj_rel     = (const float*)d_in[1];
  const int*   neigh        = (const int*)d_in[2];
  const int*   robot_idx    = (const int*)d_in[3];
  const float* r_goal       = (const float*)d_in[4];
  const float* r_pose       = (const float*)d_in[5];
  const float* action       = (const float*)d_in[6];
  const float* W_he  = (const float*)d_in[7];
  const float* b_he  = (const float*)d_in[8];
  const float* W_ih  = (const float*)d_in[9];
  const float* W_hh  = (const float*)d_in[10];
  const float* b_ih  = (const float*)d_in[11];
  const float* b_hh  = (const float*)d_in[12];
  const float* W_sp  = (const float*)d_in[13];
  const float* b_sp  = (const float*)d_in[14];
  const float* W_p1  = (const float*)d_in[15];
  const float* b_p1  = (const float*)d_in[16];
  const float* W_p2  = (const float*)d_in[17];
  const float* b_p2  = (const float*)d_in[18];
  const float* W_emb = (const float*)d_in[19];
  const float* b_emb = (const float*)d_in[20];
  const float* W_fc  = (const float*)d_in[21];
  const float* b_fc  = (const float*)d_in[22];

  _Float16* ws_hist = (_Float16*)d_ws;                                         // 48 KB
  float*    ws_pool = (float*)((char*)d_ws + NAGENTS * 16 * sizeof(_Float16)); // 96 KB
  const float* pos_last = obs_traj_pos + (TSTEPS - 1) * NAGENTS * 2;

  lstm_encoder_kernel<<<NAGENTS / 256, 256, 0, stream>>>(
      traj_rel, W_he, b_he, W_ih, W_hh, b_ih, b_hh, ws_hist);

  pooling_wmma_kernel<<<NAGENTS / 8, 256, 0, stream>>>(
      pos_last, ws_hist, neigh, W_sp, b_sp, W_p1, b_p1, W_p2, b_p2, ws_pool);

  fusion_kernel<<<NROB, 256, 0, stream>>>(
      pos_last, robot_idx, r_goal, r_pose, action, ws_pool,
      W_emb, b_emb, W_fc, b_fc, (float*)d_out);
}
